// AttDecoder_75668733821268
// MI455X (gfx1250) — matrix operand order repaired and checked by reading, not compiled, not run on hardware
//
#include <hip/hip_runtime.h>
#include <hip/hip_bf16.h>
#include <cstdint>
#include <cstddef>

// ---------------------------------------------------------------------------
// Problem constants
// ---------------------------------------------------------------------------
#define BB   16      // batch
#define TT   128     // decoder length
#define SS   512     // encoder length
#define HH   512     // hidden
#define EE   512     // embedding
#define VV   32000   // vocab
#define MROWS (BB*TT)          // 2048 GEMM rows (b*T+t)

typedef __attribute__((ext_vector_type(16))) __bf16 v16bf;
typedef __attribute__((ext_vector_type(8)))  __bf16 v8bf;
typedef __attribute__((ext_vector_type(8)))  float  v8f;

__device__ __forceinline__ float sigf(float x) { return 1.f / (1.f + __expf(-x)); }

// ---------------------------------------------------------------------------
// WMMA bf16 16x16 tile mainloop.
// A: row-major [16 x K] bf16 (lda elements).  Contiguous-k A order used here (same as Bp):
//   lanes 0-15 row M=lane, elems 0..15 = K(kb..kb+15), kb=(lane>>4)*16
// Bp: fragment-packed B: frag(kt) base + lane*16, 16 contiguous K-col elems.
// ---------------------------------------------------------------------------
__device__ __forceinline__ v8f wmma_loop(const __bf16* __restrict__ A, int lda,
                                         const __bf16* __restrict__ Bp,
                                         int ksteps, int lane) {
  v8f acc = {};
  const int r  = lane & 15;
  const int kb = (lane >> 4) << 4;
  const __bf16* arow = A + (size_t)r * lda;
  const __bf16* bl   = Bp + ((size_t)lane << 4);
#pragma unroll 4
  for (int kt = 0; kt < ksteps; ++kt) {
    const __bf16* ap = arow + kt * 32;
    v8bf alo = *(const v8bf*)(ap + kb);
    v8bf ahi = *(const v8bf*)(ap + kb + 8);
    const __bf16* bp = bl + (size_t)kt * 512;
    v8bf b0 = *(const v8bf*)(bp);
    v8bf b1 = *(const v8bf*)(bp + 8);
    if (kt + 1 < ksteps) __builtin_prefetch(bl + (size_t)(kt + 1) * 512, 0, 1);
    v16bf a, b;
#pragma unroll
    for (int i = 0; i < 8; ++i) {
      a[i] = alo[i]; a[i + 8] = ahi[i];
      b[i] = b0[i];  b[i + 8] = b1[i];
    }
    acc = __builtin_amdgcn_wmma_f32_16x16x32_bf16(false, a, false, b,
                                                  (short)0, acc, false, false);
  }
  return acc;
}

// ---------------------------------------------------------------------------
// Pack a K x N fp32 matrix (optionally transposed source, optional second
// source summed in — used to fuse the encoder direction-sum) into WMMA
// B-fragment layout: dst[(frag*32+lane)*16 + i], frag = nt*Ksteps + kt.
// ---------------------------------------------------------------------------
__global__ void pack_b_kernel(__bf16* __restrict__ dst,
                              const float* __restrict__ src,
                              const float* __restrict__ src2,
                              int Ksteps, int Ntiles, int ld, int transposed,
                              size_t src_bstride, size_t dst_bstride) {
  int tid = blockIdx.x * blockDim.x + threadIdx.x;
  int total = Ksteps * Ntiles * 32;
  if (tid >= total) return;
  int l = tid & 31, frag = tid >> 5;
  int kt = frag % Ksteps, nt = frag / Ksteps;
  int col = nt * 16 + (l & 15);
  int k0  = kt * 32 + ((l >> 4) << 4);
  size_t boff = (size_t)blockIdx.y * src_bstride;
  const float* s  = src + boff;
  const float* s2 = src2 ? (src2 + boff) : nullptr;
  __bf16* d = dst + (size_t)blockIdx.y * dst_bstride + ((size_t)frag * 32 + l) * 16;
#pragma unroll 4
  for (int i = 0; i < 16; ++i) {
    size_t si = transposed ? ((size_t)col * ld + (k0 + i))
                           : ((size_t)(k0 + i) * ld + col);
    float v = s[si];
    if (s2) v += s2[si];
    d[i] = (__bf16)v;
  }
}

// ---------------------------------------------------------------------------
// Generic (optionally batched) bf16 GEMM: one wave per 16x16 C tile.
// ---------------------------------------------------------------------------
__global__ void gemm_bf16_kernel(const __bf16* __restrict__ A, int lda, size_t a_bs,
                                 const __bf16* __restrict__ Bp, size_t b_bs,
                                 const float* __restrict__ bias,
                                 float* __restrict__ C, int ldc, size_t c_bs,
                                 int Mtiles, int Ntiles, int Ksteps) {
  int lane = threadIdx.x & 31;
  int wid  = blockIdx.x * (blockDim.x >> 5) + (threadIdx.x >> 5);
  if (wid >= Mtiles * Ntiles) return;   // uniform per wave -> EXEC stays all-ones
  int mt = wid % Mtiles, nt = wid / Mtiles;
  size_t bz = blockIdx.y;
  const __bf16* Ab = A + bz * a_bs + (size_t)mt * 16 * lda;
  const __bf16* Bb = Bp + bz * b_bs + (size_t)nt * Ksteps * 512;
  v8f acc = wmma_loop(Ab, lda, Bb, Ksteps, lane);
  int m0  = mt * 16 + ((lane >> 4) << 3);
  int col = nt * 16 + (lane & 15);
  float bv = bias ? bias[col] : 0.f;
  float* Cb = C + bz * c_bs;
#pragma unroll
  for (int r = 0; r < 8; ++r)
    Cb[(size_t)(m0 + r) * ldc + col] = acc[r] + bv;
}

// ---------------------------------------------------------------------------
// Align GEMM with fused epilogue:
//   aligned = sigmoid(acc + b_align); out_pre = lstm_out + h_bar * aligned (bf16)
// ---------------------------------------------------------------------------
__global__ void align_gemm_kernel(const __bf16* __restrict__ A,
                                  const __bf16* __restrict__ Bp,
                                  const float* __restrict__ b_align,
                                  const float* __restrict__ lstm_out,
                                  const float* __restrict__ hbar,
                                  __bf16* __restrict__ outpre) {
  const int Mtiles = MROWS / 16, Ntiles = HH / 16, Ksteps = (2 * HH) / 32;
  int lane = threadIdx.x & 31;
  int wid  = blockIdx.x * (blockDim.x >> 5) + (threadIdx.x >> 5);
  if (wid >= Mtiles * Ntiles) return;
  int mt = wid % Mtiles, nt = wid / Mtiles;
  const __bf16* Ab = A + (size_t)mt * 16 * (2 * HH);
  const __bf16* Bb = Bp + (size_t)nt * Ksteps * 512;
  v8f acc = wmma_loop(Ab, 2 * HH, Bb, Ksteps, lane);
  int m0  = mt * 16 + ((lane >> 4) << 3);
  int col = nt * 16 + (lane & 15);
#pragma unroll
  for (int r = 0; r < 8; ++r) {
    float al = sigf(acc[r] + b_align[col]);
    size_t idx = (size_t)(m0 + r) * HH + col;
    outpre[idx] = (__bf16)(lstm_out[idx] + hbar[idx] * al);
  }
}

// ---------------------------------------------------------------------------
// Persistent single-workgroup LSTM over T=128 steps (sequential recurrence).
// 32 waves; per step each wave does 4 N-tiles of gates = h @ W_hh^T (+gates_x),
// then 1024 threads apply gate nonlinearities + packed-sequence masking.
// LDS: h_f32(32K) + c_f32(32K) + gates(128K) + h_bf16(16K) = 208 KB.
// ---------------------------------------------------------------------------
__global__ void lstm_seq_kernel(const int* __restrict__ x_len,
                                const float* __restrict__ h0,
                                const float* __restrict__ c0,
                                const float* __restrict__ gates_x,
                                const __bf16* __restrict__ whh_pack,
                                float* __restrict__ lstm_out) {
  extern __shared__ char smem[];
  float*  h_f   = (float*)smem;                    // [16*512]
  float*  c_f   = h_f + BB * HH;                   // [16*512]
  float*  gates = c_f + BB * HH;                   // [16*2048]
  __bf16* h_b   = (__bf16*)(gates + BB * 4 * HH);  // [16*512]

  const int tid  = threadIdx.x;
  const int lane = tid & 31;
  const int w    = tid >> 5;

  for (int j = tid; j < BB * HH; j += blockDim.x) {
    float h = h0[j], c = c0[j];
    h_f[j] = h; c_f[j] = c; h_b[j] = (__bf16)h;
  }
  __syncthreads();

  for (int t = 0; t < TT; ++t) {
    // gates[16][2048] = h @ W_hh^T + gates_x[row(b,t)]
#pragma unroll
    for (int nti = 0; nti < 4; ++nti) {
      int nt = w * 4 + nti;
      v8f acc = wmma_loop(h_b, HH, whh_pack + (size_t)nt * 16 * 512, HH / 32, lane);
      int m0  = (lane >> 4) << 3;
      int col = nt * 16 + (lane & 15);
#pragma unroll
      for (int r = 0; r < 8; ++r) {
        int bidx = m0 + r;
        gates[bidx * (4 * HH) + col] =
            acc[r] + gates_x[((size_t)bidx * TT + t) * (4 * HH) + col];
      }
    }
    __syncthreads();

    for (int j = tid; j < BB * HH; j += blockDim.x) {
      int b = j >> 9, u = j & (HH - 1);
      const float* g = gates + b * (4 * HH);
      float i_ = sigf(g[u]);
      float f_ = sigf(g[HH + u]);
      float gg = tanhf(g[2 * HH + u]);
      float o_ = sigf(g[3 * HH + u]);
      float cn = f_ * c_f[j] + i_ * gg;
      float hn = o_ * tanhf(cn);
      bool valid = (t < x_len[b]);
      float h2 = valid ? hn : h_f[j];
      float c2 = valid ? cn : c_f[j];
      h_f[j] = h2; c_f[j] = c2; h_b[j] = (__bf16)h2;
      lstm_out[((size_t)b * TT + t) * HH + u] = valid ? hn : 0.f;
    }
    __syncthreads();
  }
}

// ---------------------------------------------------------------------------
// Elementwise / build kernels
// ---------------------------------------------------------------------------
__global__ void build_lstm_input_kernel(const int* __restrict__ x,
                                        const float* __restrict__ h0,
                                        const float* __restrict__ c0,
                                        const float* __restrict__ emb,
                                        float* __restrict__ li_f,
                                        __bf16* __restrict__ li_b) {
  size_t idx = (size_t)blockIdx.x * blockDim.x + threadIdx.x;
  if (idx >= (size_t)MROWS * (HH + EE)) return;
  int m = (int)(idx >> 10), col = (int)(idx & 1023);
  int b = m >> 7;
  float v;
  if (col < HH) v = h0[b * HH + col] + c0[b * HH + col];       // context
  else          v = emb[(size_t)x[m] * EE + (col - HH)];        // embedding
  li_f[idx] = v; li_b[idx] = (__bf16)v;
}

__global__ void build_attn_in_kernel(const float* __restrict__ lstm_out,
                                     const float* __restrict__ li_f,
                                     float* __restrict__ attn_f,
                                     __bf16* __restrict__ attn_b) {
  size_t idx = (size_t)blockIdx.x * blockDim.x + threadIdx.x;
  if (idx >= (size_t)MROWS * 1536) return;
  int m = (int)(idx / 1536), col = (int)(idx % 1536);
  float v = (col < HH) ? lstm_out[(size_t)m * HH + col]
                       : li_f[(size_t)m * 1024 + (col - HH)];
  attn_f[idx] = v; attn_b[idx] = (__bf16)v;
}

__global__ void build_align_in_kernel(const float* __restrict__ lstm_out,
                                      const float* __restrict__ hbar,
                                      __bf16* __restrict__ dst) {
  size_t idx = (size_t)blockIdx.x * blockDim.x + threadIdx.x;
  if (idx >= (size_t)MROWS * 1024) return;
  int m = (int)(idx >> 10), col = (int)(idx & 1023);
  float v = (col < HH) ? lstm_out[(size_t)m * HH + col]
                       : hbar[(size_t)m * HH + (col - HH)];
  dst[idx] = (__bf16)v;
}

__global__ void bias_sum_kernel(const float* a, const float* b, float* o, int n) {
  int i = blockIdx.x * blockDim.x + threadIdx.x;
  if (i < n) o[i] = a[i] + b[i];
}

// Softmax over S=512: one wave per row, 16 elems/lane, shuffle reductions.
__global__ void softmax_kernel(const float* __restrict__ logits,
                               __bf16* __restrict__ soft) {
  int row  = blockIdx.x * 8 + (threadIdx.x >> 5);
  int lane = threadIdx.x & 31;
  if (row >= MROWS) return;
  const float* lr = logits + (size_t)row * SS;
  float v[16], mx = -3.4e38f;
#pragma unroll
  for (int i = 0; i < 16; ++i) { v[i] = lr[lane + i * 32]; mx = fmaxf(mx, v[i]); }
  for (int off = 16; off > 0; off >>= 1) mx = fmaxf(mx, __shfl_xor(mx, off, 32));
  float sum = 0.f;
#pragma unroll
  for (int i = 0; i < 16; ++i) { v[i] = __expf(v[i] - mx); sum += v[i]; }
  for (int off = 16; off > 0; off >>= 1) sum += __shfl_xor(sum, off, 32);
  float inv = 1.f / sum;
  __bf16* sr = soft + (size_t)row * SS;
#pragma unroll
  for (int i = 0; i < 16; ++i) sr[lane + i * 32] = (__bf16)(v[i] * inv);
}

__global__ void targets_kernel(int* __restrict__ tgt) {
  int i = blockIdx.x * blockDim.x + threadIdx.x;
  if (i >= BB * TT) return;
  int t = i & (TT - 1);
  tgt[i] = (SS / TT) * t;   // t=0 -> 0 already
}

// ---------------------------------------------------------------------------
// Host orchestration
// ---------------------------------------------------------------------------
extern "C" void kernel_launch(void* const* d_in, const int* in_sizes, int n_in,
                              void* d_out, int out_size, void* d_ws, size_t ws_size,
                              hipStream_t stream) {
  const int*   x     = (const int*)d_in[0];
  const int*   xlen  = (const int*)d_in[1];
  const float* h0    = (const float*)d_in[2];
  const float* c0    = (const float*)d_in[3];
  const float* enc_o = (const float*)d_in[4];
  const float* emb   = (const float*)d_in[5];
  const float* W_att = (const float*)d_in[6];
  const float* b_att = (const float*)d_in[7];
  const float* W_ih  = (const float*)d_in[8];
  const float* W_hh  = (const float*)d_in[9];
  const float* b_ih  = (const float*)d_in[10];
  const float* b_hh  = (const float*)d_in[11];
  const float* W_al  = (const float*)d_in[12];
  const float* b_al  = (const float*)d_in[13];
  const float* W_out = (const float*)d_in[14];
  const float* b_out = (const float*)d_in[15];

  float* out_logits = (float*)d_out;                               // [B,T,V]
  float* out_attn   = out_logits + (size_t)MROWS * VV;             // [B,T,1536]
  int*   out_tgt    = (int*)(out_attn + (size_t)MROWS * 1536);     // [B,T]

  // workspace carve-up
  char* ws = (char*)d_ws; size_t off = 0;
  auto carve = [&](size_t bytes) -> void* {
    void* p = ws + off; off = (off + bytes + 255) & ~(size_t)255; return p;
  };
  float*  li_f     = (float*) carve((size_t)MROWS * 1024 * 4);
  __bf16* li_b     = (__bf16*)carve((size_t)MROWS * 1024 * 2);
  __bf16* wih_p    = (__bf16*)carve((size_t)32 * 128 * 512 * 2);
  __bf16* whh_p    = (__bf16*)carve((size_t)16 * 128 * 512 * 2);
  __bf16* watt_p   = (__bf16*)carve((size_t)48 * 32 * 512 * 2);
  __bf16* wal_p    = (__bf16*)carve((size_t)32 * 32 * 512 * 2);
  __bf16* wout_p   = (__bf16*)carve((size_t)16 * 2000 * 512 * 2);
  __bf16* enc_p    = (__bf16*)carve((size_t)BB * 16 * 32 * 512 * 2);
  float*  gates_x  = (float*) carve((size_t)MROWS * 2048 * 4);
  float*  bsum     = (float*) carve(2048 * 4);
  float*  lstm_f   = (float*) carve((size_t)MROWS * HH * 4);
  __bf16* attn_b   = (__bf16*)carve((size_t)MROWS * 1536 * 2);
  float*  logits   = (float*) carve((size_t)MROWS * SS * 4);
  __bf16* soft_b   = (__bf16*)carve((size_t)MROWS * SS * 2);
  float*  hbar_f   = (float*) carve((size_t)MROWS * HH * 4);
  __bf16* alin_b   = (__bf16*)carve((size_t)MROWS * 1024 * 2);
  __bf16* outpre_b = (__bf16*)carve((size_t)MROWS * HH * 2);

  const int TPB = 256;
  auto blks = [](size_t n, int t) { return (unsigned)((n + t - 1) / t); };

  // --- weight packing (bf16 fragment layout) ---
  pack_b_kernel<<<blks(32 * 128 * 32, TPB), TPB, 0, stream>>>(
      wih_p, W_ih, nullptr, 32, 128, 1024, 1, 0, 0);          // W_ih^T: K=1024,N=2048
  pack_b_kernel<<<blks(16 * 128 * 32, TPB), TPB, 0, stream>>>(
      whh_p, W_hh, nullptr, 16, 128, 512, 1, 0, 0);           // W_hh^T: K=512, N=2048
  pack_b_kernel<<<blks(48 * 32 * 32, TPB), TPB, 0, stream>>>(
      watt_p, W_att, nullptr, 48, 32, 512, 0, 0, 0);          // K=1536,N=512
  pack_b_kernel<<<blks(32 * 32 * 32, TPB), TPB, 0, stream>>>(
      wal_p, W_al, nullptr, 32, 32, 512, 0, 0, 0);            // K=1024,N=512
  pack_b_kernel<<<blks(16 * 2000 * 32, TPB), TPB, 0, stream>>>(
      wout_p, W_out, nullptr, 16, 2000, 32000, 0, 0, 0);      // K=512, N=32000
  {   // enc = dir-sum of encoder_outputs, packed per batch (flat (D,B,S,H) view)
    dim3 g(blks(16 * 32 * 32, TPB), BB);
    pack_b_kernel<<<g, TPB, 0, stream>>>(
        enc_p, enc_o, enc_o + (size_t)BB * SS * HH, 16, 32, HH, 0,
        (size_t)SS * HH, (size_t)16 * 32 * 512);
  }
  bias_sum_kernel<<<blks(2048, TPB), TPB, 0, stream>>>(b_ih, b_hh, bsum, 2048);

  // --- lstm_input = [context | embedding] ---
  build_lstm_input_kernel<<<blks((size_t)MROWS * 1024, TPB), TPB, 0, stream>>>(
      x, h0, c0, emb, li_f, li_b);

  // --- gates_x = lstm_input @ W_ih^T + (b_ih+b_hh)  (hoisted, parallel) ---
  gemm_bf16_kernel<<<blks((size_t)128 * 128, 8), TPB, 0, stream>>>(
      li_b, 1024, 0, wih_p, 0, bsum, gates_x, 2048, 0, 128, 128, 32);

  // --- sequential LSTM: one persistent workgroup ---
  lstm_seq_kernel<<<1, 1024, 212992, stream>>>(xlen, h0, c0, gates_x, whh_p, lstm_f);

  // --- attention_input (also an output) + logits + softmax + h_bar ---
  build_attn_in_kernel<<<blks((size_t)MROWS * 1536, TPB), TPB, 0, stream>>>(
      lstm_f, li_f, out_attn, attn_b);
  gemm_bf16_kernel<<<blks((size_t)128 * 32, 8), TPB, 0, stream>>>(
      attn_b, 1536, 0, watt_p, 0, b_att, logits, 512, 0, 128, 32, 48);
  softmax_kernel<<<blks(MROWS, 8), TPB, 0, stream>>>(logits, soft_b);
  {   // h_bar[b] = soft[b] @ enc[b]   (batched over 16)
    dim3 g(blks((size_t)8 * 32, 8), BB);
    gemm_bf16_kernel<<<g, TPB, 0, stream>>>(
        soft_b, 512, (size_t)TT * 512, enc_p, (size_t)16 * 32 * 512, nullptr,
        hbar_f, 512, (size_t)TT * 512, 8, 32, 16);
  }

  // --- align gate + combine -> out_pre (bf16) ---
  build_align_in_kernel<<<blks((size_t)MROWS * 1024, TPB), TPB, 0, stream>>>(
      lstm_f, hbar_f, alin_b);
  align_gemm_kernel<<<blks((size_t)128 * 32, 8), TPB, 0, stream>>>(
      alin_b, wal_p, b_al, lstm_f, hbar_f, outpre_b);

  // --- final projection: [2048x512] @ [512x32000] + b_out -> d_out ---
  gemm_bf16_kernel<<<blks((size_t)128 * 2000, 8), TPB, 0, stream>>>(
      outpre_b, 512, 0, wout_p, 0, b_out, out_logits, VV, 0, 128, 2000, 16);

  // --- attention targets ---
  targets_kernel<<<blks(BB * TT, TPB), TPB, 0, stream>>>(out_tgt);
}
